// ChildSumLSTMCell_51891794870896
// MI455X (gfx1250) — matrix operand order repaired and checked by reading, not compiled, not run on hardware
//
#include <hip/hip_runtime.h>
#include <hip/hip_bf16.h>

// ---------------------------------------------------------------------------
// ChildSum Tree-LSTM for MI455X (gfx1250, wave32, WMMA, async LDS loads).
// ---------------------------------------------------------------------------

#define N_NODES 16384
#define HID     512
#define NGATE   2048   // 4*HID

typedef __attribute__((ext_vector_type(16))) __bf16 v16bf;
typedef __attribute__((ext_vector_type(8)))  float  v8f;
typedef int v4i __attribute__((vector_size(16)));   // matches builtin param type

#define AS1 __attribute__((address_space(1)))
#define AS3 __attribute__((address_space(3)))

#if __has_builtin(__builtin_amdgcn_global_load_async_to_lds_b128)
#define HAVE_ASYNC_LDS 1
#else
#define HAVE_ASYNC_LDS 0
#endif

__device__ __forceinline__ void async_copy_b128(const unsigned short* g,
                                                unsigned short* l) {
#if HAVE_ASYNC_LDS
    __builtin_amdgcn_global_load_async_to_lds_b128((AS1 v4i*)g, (AS3 v4i*)l,
                                                   0, 0);
#else
    *(uint4*)l = *(const uint4*)g;
#endif
}

__device__ __forceinline__ void async_wait0() {
#if HAVE_ASYNC_LDS
#if __has_builtin(__builtin_amdgcn_s_wait_asynccnt)
    __builtin_amdgcn_s_wait_asynccnt(0);
#else
    asm volatile("s_wait_asynccnt 0x0" ::: "memory");
#endif
#endif
}

__device__ __forceinline__ unsigned short f2bf(float x) {
    union { float f; unsigned u; } v; v.f = x;
    unsigned r = v.u + 0x7FFFu + ((v.u >> 16) & 1u);   // round-to-nearest-even
    return (unsigned short)(r >> 16);
}

__device__ __forceinline__ float sigmoidf(float x) {
    return 1.0f / (1.0f + __expf(-x));
}

// --------------------------- fp32 -> bf16 conversion -----------------------
__global__ void cvt_bf16(const float* __restrict__ src,
                         unsigned short* __restrict__ dst, int n) {
    int i = blockIdx.x * blockDim.x + threadIdx.x;
    if (i < n) dst[i] = f2bf(src[i]);
}

// fp32 [R,C] row-major -> bf16 [C,R] (transpose): one-time weight prep.
__global__ void cvt_bf16_T(const float* __restrict__ src,
                           unsigned short* __restrict__ dst, int R, int C) {
    int i = blockIdx.x * blockDim.x + threadIdx.x;
    if (i < R * C) {
        int r = i / C, c = i % C;
        dst[(size_t)c * R + r] = f2bf(src[i]);
    }
}

// --------------------------- i2h WMMA GEMM ---------------------------------
// C[M,N] = A[M,K](bf16) * W[N,K]^T(bf16) + bias[N], fp32 out.
// 256 threads = 8 waves. Block tile 32(M) x 128(N), K-stage 64.
// Double-buffered LDS, async global->LDS loads overlapped with WMMA.
#define BM 32
#define BN 128
#define BK 64

__global__ void __launch_bounds__(256)
gemm_i2h(const unsigned short* __restrict__ A,
         const unsigned short* __restrict__ W,
         const float* __restrict__ bias,
         float* __restrict__ Cout,
         int M, int N, int K) {
    __shared__ unsigned short sA[2][BM * BK];   // 2 x 4 KB
    __shared__ unsigned short sB[2][BN * BK];   // 2 x 16 KB

    const int tid  = threadIdx.x;
    const int wave = tid >> 5;
    const int lane = tid & 31;
    const int m0   = blockIdx.y * BM;
    const int n0b  = blockIdx.x * BN;

    auto stage = [&](int buf, int kk) {
        // A tile: 32x64 bf16 = 256 b128 transfers, one per thread
        {
            int r = tid >> 3, c = (tid & 7) * 8;
            async_copy_b128(A + (size_t)(m0 + r) * K + kk + c,
                            &sA[buf][r * BK + c]);
        }
        // B tile: 128x64 bf16 = 1024 b128 transfers, four per thread
        #pragma unroll
        for (int i = 0; i < 4; ++i) {
            int idx = tid + 256 * i;
            int r = idx >> 3, c = (idx & 7) * 8;
            async_copy_b128(W + (size_t)(n0b + r) * K + kk + c,
                            &sB[buf][r * BK + c]);
        }
    };

    v8f acc0 = {}, acc1 = {};
    const int mrow = lane & 15;
    const int kbA  = (lane < 16) ? 0 : 16;
    const int ncol = wave * 16 + (lane & 15);
    const int kbB  = (lane < 16) ? 0 : 16;

    stage(0, 0);
    async_wait0();
    __syncthreads();

    const int nstage = K / BK;
    for (int s = 0; s < nstage; ++s) {
        const int buf = s & 1;
        if (s + 1 < nstage) stage(buf ^ 1, (s + 1) * BK);   // overlap with compute

        #pragma unroll
        for (int ks = 0; ks < 2; ++ks) {                    // two k-steps per stage
            const int kk = ks * 32;
            union { v16bf v; unsigned u[8]; } bf_;
            #pragma unroll
            for (int v = 0; v < 8; ++v)
                bf_.u[v] = *(const unsigned*)&sB[buf][ncol * BK + kk + kbB + 2 * v];
            #pragma unroll
            for (int mt = 0; mt < 2; ++mt) {                // two M tiles per wave
                union { v16bf v; unsigned u[8]; } af;
                #pragma unroll
                for (int v = 0; v < 8; ++v) {
                    int k = kbA + 2 * v;
                    af.u[v] = *(const unsigned*)&sA[buf][(mt * 16 + mrow) * BK + kk + k];
                }
                if (mt == 0)
                    acc0 = __builtin_amdgcn_wmma_f32_16x16x32_bf16(
                               false, af.v, false, bf_.v, (short)0, acc0, false, false);
                else
                    acc1 = __builtin_amdgcn_wmma_f32_16x16x32_bf16(
                               false, af.v, false, bf_.v, (short)0, acc1, false, false);
            }
        }
        async_wait0();
        __syncthreads();
    }

    #pragma unroll
    for (int r = 0; r < 8; ++r) {
        int Mr = r + 8 * (lane >> 4);
        int Nc = n0b + wave * 16 + (lane & 15);
        Cout[(size_t)(m0 + Mr) * N + Nc]      = acc0[r] + bias[Nc];
        Cout[(size_t)(m0 + 16 + Mr) * N + Nc] = acc1[r] + bias[Nc];
    }
}

// --------------------------- sequential tree scan --------------------------
// Single persistent workgroup: 1024 threads = 32 waves.
// H/C state lives directly in d_out (children always precede parents).
__global__ void __launch_bounds__(1024)
scan_kernel(const float* __restrict__ i2h,           // [N_NODES, 2048] fp32
            const unsigned short* __restrict__ Wct,  // [512,512]  bf16, TRANSPOSED: [n][k]
            const unsigned short* __restrict__ Whs,  // [1536,512] bf16: [n][k]
            const float* __restrict__ hs2h_bias,     // [1536]
            const float* __restrict__ hc2h_bias,     // [512]
            const int*   __restrict__ child_idx,     // [N,4]
            const int*   __restrict__ child_mask,    // [N,4]
            float* __restrict__ Hst,                 // [N,512]
            float* __restrict__ Cst,                 // [N,512]
            int n_nodes) {
    __shared__ unsigned short sHc[16][HID];  // 16 KB  A-tile: rows 0..3 children h, 4..15 zero
    __shared__ float          sHf[4][HID];   //  8 KB  children h fp32
    __shared__ float          sCc[4][HID];   //  8 KB  children c fp32
    __shared__ unsigned short sHs[HID];      //  1 KB  hs (bf16), row 0 of hs2h A-tile
    __shared__ float          sF [4][HID];   //  8 KB  per-child forget gates
    __shared__ float          sIUO[3 * HID]; //  6 KB  i/u/o pre-activations

    const int tid  = threadIdx.x;
    const int wave = tid >> 5;
    const int lane = tid & 31;

    for (int e = tid; e < 12 * HID; e += 1024)          // rows 4..15 stay zero
        sHc[4 + (e >> 9)][e & 511] = 0;
    __syncthreads();

    for (int node = 0; node < n_nodes; ++node) {
        int ci[4], cm[4];
        #pragma unroll
        for (int j = 0; j < 4; ++j) {
            ci[j] = child_idx[node * 4 + j];
            cm[j] = child_mask[node * 4 + j];
        }

        // prefetch next node's i2h row (8 KB = 64 cachelines) behind this step
        if (node + 1 < n_nodes && tid < 64)
            __builtin_prefetch(i2h + (size_t)(node + 1) * NGATE + tid * 32, 0, 1);

        // ---- phase 1: gather children H/C into LDS ----
        #pragma unroll
        for (int rep = 0; rep < 2; ++rep) {
            int e  = tid + rep * 1024;
            int ch = e >> 9, c = e & 511;
            float hv = Hst[(size_t)ci[ch] * HID + c];
            float cv = Cst[(size_t)ci[ch] * HID + c];
            sHf[ch][c] = hv;
            sCc[ch][c] = cv;
            sHc[ch][c] = f2bf(hv);
        }
        __syncthreads();
        if (tid < HID) {
            float s = 0.f;
            #pragma unroll
            for (int j = 0; j < 4; ++j) s += cm[j] ? sHf[j][tid] : 0.f;
            sHs[tid] = f2bf(s);
        }
        __syncthreads();

        // ---- phase 2a: f = sigmoid(f_pre + b + hc @ Whc); wave w -> cols [16w,16w+16)
        {
            v8f acc = {};
            const int mrow = lane & 15;
            const int kbA  = (lane < 16) ? 0 : 16;
            const int ncol = wave * 16 + (lane & 15);
            for (int kk = 0; kk < HID; kk += 32) {
                union { v16bf v; unsigned u[8]; } af, bf_;
                #pragma unroll
                for (int v = 0; v < 8; ++v) {
                    int k = kbA + 2 * v;
                    af.u[v] = *(const unsigned*)&sHc[mrow][kk + k];
                }
                const int kbB = kk + ((lane < 16) ? 0 : 16);
                #pragma unroll
                for (int v = 0; v < 8; ++v)   // B[k][n] = Wct[n][k]: contiguous pair
                    bf_.u[v] = *(const unsigned*)&Wct[(size_t)ncol * HID + kbB + 2 * v];
                acc = __builtin_amdgcn_wmma_f32_16x16x32_bf16(
                          false, af.v, false, bf_.v, (short)0, acc, false, false);
            }
            #pragma unroll
            for (int r = 0; r < 8; ++r) {
                int Mr = r + 8 * (lane >> 4);
                if (Mr < 4) {
                    int Nc = wave * 16 + (lane & 15);
                    float pre = i2h[(size_t)node * NGATE + HID + Nc]
                              + hc2h_bias[Nc] + acc[r];
                    sF[Mr][Nc] = sigmoidf(pre);
                }
            }
        }

        // ---- phase 2b: iuo = iuo_pre + hs @ Whs^T + b; wave w -> 3 tiles of 16
        #pragma unroll
        for (int t = 0; t < 3; ++t) {
            const int n0 = wave * 48 + t * 16;
            v8f acc = {};
            const bool row0 = (lane & 15) == 0;
            const int  kbA  = (lane < 16) ? 0 : 16;
            const int  ncol = n0 + (lane & 15);
            for (int kk = 0; kk < HID; kk += 32) {
                union { v16bf v; unsigned u[8]; } af, bf_;
                #pragma unroll
                for (int v = 0; v < 8; ++v) {
                    int k = kbA + 2 * v;
                    af.u[v] = row0 ? *(const unsigned*)&sHs[kk + k] : 0u;
                }
                const int kbB = kk + ((lane < 16) ? 0 : 16);
                #pragma unroll
                for (int v = 0; v < 8; ++v)
                    bf_.u[v] = *(const unsigned*)&Whs[(size_t)ncol * HID + kbB + 2 * v];
                acc = __builtin_amdgcn_wmma_f32_16x16x32_bf16(
                          false, af.v, false, bf_.v, (short)0, acc, false, false);
            }
            if (lane < 16) {
                int j   = n0 + lane;                  // 0..1535 in (i,u,o) order
                int col = (j < HID) ? j : (j + HID);  // map into i2h's [i,f,u,o]
                sIUO[j] = acc[0] + i2h[(size_t)node * NGATE + col] + hs2h_bias[j];
            }
        }
        __syncthreads();

        // ---- phase 3: gates + state update ----
        if (tid < HID) {
            int c = tid;
            float fc = 0.f;
            #pragma unroll
            for (int j = 0; j < 4; ++j)
                fc += cm[j] ? sF[j][c] * sCc[j][c] : 0.f;
            float ig = sIUO[c], ug = sIUO[HID + c], og = sIUO[2 * HID + c];
            float nc = sigmoidf(ig) * tanhf(ug) + fc;
            float nh = sigmoidf(og) * tanhf(nc);
            Hst[(size_t)node * HID + c] = nh;
            Cst[(size_t)node * HID + c] = nc;
        }
        __syncthreads();
    }
}

// ---------------------------------------------------------------------------
extern "C" void kernel_launch(void* const* d_in, const int* in_sizes, int n_in,
                              void* d_out, int out_size, void* d_ws, size_t ws_size,
                              hipStream_t stream) {
    const float* inputs    = (const float*)d_in[0];
    const float* i2h_w     = (const float*)d_in[1];
    const float* i2h_b     = (const float*)d_in[2];
    const float* hs2h_w    = (const float*)d_in[3];
    const float* hs2h_b    = (const float*)d_in[4];
    const float* hc2h_w    = (const float*)d_in[5];
    const float* hc2h_b    = (const float*)d_in[6];
    const int*   child_idx = (const int*)d_in[7];
    const int*   child_mask= (const int*)d_in[8];
    float* out = (float*)d_out;

    char* ws = (char*)d_ws;
    float* i2h = (float*)ws;                                   // 128 MB
    size_t off = (size_t)N_NODES * NGATE * sizeof(float);
    unsigned short* Ab  = (unsigned short*)(ws + off); off += (size_t)N_NODES * HID * 2;
    unsigned short* Wib = (unsigned short*)(ws + off); off += (size_t)NGATE * HID * 2;
    unsigned short* Wsb = (unsigned short*)(ws + off); off += (size_t)3 * HID * HID * 2;
    unsigned short* Wct = (unsigned short*)(ws + off); off += (size_t)HID * HID * 2;

    int n;
    n = N_NODES * HID;  cvt_bf16<<<(n + 255) / 256, 256, 0, stream>>>(inputs, Ab, n);
    n = NGATE * HID;    cvt_bf16<<<(n + 255) / 256, 256, 0, stream>>>(i2h_w,  Wib, n);
    n = 3 * HID * HID;  cvt_bf16<<<(n + 255) / 256, 256, 0, stream>>>(hs2h_w, Wsb, n);
    n = HID * HID;      cvt_bf16_T<<<(n + 255) / 256, 256, 0, stream>>>(hc2h_w, Wct, HID, HID);

    dim3 grid(NGATE / BN, N_NODES / BM);
    gemm_i2h<<<grid, 256, 0, stream>>>(Ab, Wib, i2h_b, i2h, N_NODES, NGATE, HID);

    float* Hst = out;
    float* Cst = out + (size_t)N_NODES * HID;
    scan_kernel<<<1, 1024, 0, stream>>>(i2h, Wct, Wsb, hs2h_b, hc2h_b,
                                        child_idx, child_mask, Hst, Cst, N_NODES);
}
